// TDENTransformerLayer_57844619542564
// MI455X (gfx1250) — matrix operand order repaired and checked by reading, not compiled, not run on hardware
//
#include <hip/hip_runtime.h>
#include <math.h>

typedef __attribute__((ext_vector_type(16))) _Float16 v16h;
typedef __attribute__((ext_vector_type(8)))  _Float16 v8h;
typedef __attribute__((ext_vector_type(8)))  float    v8f;

#define N_TOK   65536
#define IMG     256
#define C_DIM   192
#define C3      576
#define M_TOK   128
#define RC      20
#define HID     384
#define WSZ     16
#define SHIFT_  8
#define HEADS_  6
#define HD      32
#define NN_     256
#define LOG_M   4.8520302639f   // ln(128)
#define QSCALE  0.1767766953f   // 32^-0.5

// ---------------------------------------------------------------- WMMA fragment order
// A (16x32 f16): lane<16 -> K base 0, lane>=16 -> K base 8;
// element j=0..7 -> k=base+j, j=8..15 -> k=base+16+(j-8).
__device__ __forceinline__ int a_k_of(int lane, int j) {
  int base = (lane < 16) ? 0 : 8;
  return base + ((j < 8) ? j : (16 + (j - 8)));
}
// B (32x16 f16): k order as A (elem j of lane l = K a_k_of(l, j)). N = lane&15.

__device__ __forceinline__ v8f wmma_f16(v16h a, v16h b, v8f c) {
  return __builtin_amdgcn_wmma_f32_16x16x32_f16(false, a, false, b, (short)0, c, false, false);
}

__device__ __forceinline__ float gelu_exact(float v) {
  return 0.5f * v * (1.0f + erff(v * 0.70710678118f));
}

// ---------------------------------------------------------------- operand packing
__global__ void k_packA(const float* __restrict__ A, int mt, int kbc, int lda,
                        _Float16* __restrict__ out) {
  size_t idx = (size_t)blockIdx.x * 256 + threadIdx.x;
  size_t total = (size_t)kbc * mt * 512;
  if (idx >= total) return;
  int j = idx & 15;
  int lane = (idx >> 4) & 31;
  size_t rem = idx >> 9;
  int tm = (int)(rem % mt);
  int kb = (int)(rem / mt);
  int m = tm * 16 + (lane & 15);
  int k = kb * 32 + a_k_of(lane, j);
  out[idx] = (_Float16)A[(size_t)m * lda + k];
}

__global__ void k_packB(const float* __restrict__ B, int kbc, int N, int ntp,
                        _Float16* __restrict__ out) {
  size_t idx = (size_t)blockIdx.x * 256 + threadIdx.x;
  size_t total = (size_t)kbc * ntp * 512;
  if (idx >= total) return;
  int j = idx & 15;
  int lane = (idx >> 4) & 31;
  size_t rem = idx >> 9;
  int tn = (int)(rem % ntp);
  int kb = (int)(rem / ntp);
  int n = tn * 16 + (lane & 15);
  int k = kb * 32 + a_k_of(lane, j);
  out[idx] = (n < N) ? (_Float16)B[(size_t)k * N + n] : (_Float16)0.f;
}

// P-pack: A[m][k] = exp(sim[k][m]*inv + sh - mx) in A-fragment order.
__global__ void k_packP(const float* __restrict__ sim, const float* __restrict__ st,
                        _Float16* __restrict__ out) {
  size_t idx = (size_t)blockIdx.x * 256 + threadIdx.x;
  size_t total = (size_t)(N_TOK / 32) * (M_TOK / 16) * 512;
  if (idx >= total) return;
  int j = idx & 15;
  int lane = (idx >> 4) & 31;
  size_t rem = idx >> 9;
  int tm = (int)(rem % (M_TOK / 16));
  int kb = (int)(rem / (M_TOK / 16));
  int m = tm * 16 + (lane & 15);
  int k = kb * 32 + a_k_of(lane, j);
  float v = expf(sim[(size_t)k * M_TOK + m] * st[m] + st[M_TOK + m] - st[2 * M_TOK + m]);
  out[idx] = (_Float16)v;
}

// ---------------------------------------------------------------- LayerNorm: wave per token
__global__ __launch_bounds__(256) void k_layernorm(const float* __restrict__ x,
    const float* __restrict__ g, const float* __restrict__ b,
    float* __restrict__ out, int n, int c) {
  int wave = threadIdx.x >> 5, lane = threadIdx.x & 31;
  int t = blockIdx.x * 8 + wave;
  if (t >= n) return;
  const float* row = x + (size_t)t * c;
  float s = 0.f, ss = 0.f;
  for (int i = lane; i < c; i += 32) { float v = row[i]; s += v; ss += v * v; }
#pragma unroll
  for (int o = 16; o > 0; o >>= 1) { s += __shfl_xor(s, o, 32); ss += __shfl_xor(ss, o, 32); }
  float mu = s / c;
  float inv = rsqrtf(ss / c - mu * mu + 1e-5f);
  float* orow = out + (size_t)t * c;
  for (int i = lane; i < c; i += 32) orow[i] = (row[i] - mu) * inv * g[i] + b[i];
}

// ---------------------------------------------------------------- packed WMMA GEMM
// Wave computes 2 M-tiles x 2 N-tiles (software pipelined). Requires mt, ntp even.
__global__ __launch_bounds__(64) void k_gemm(const _Float16* __restrict__ Ap,
    const _Float16* __restrict__ Bp, const float* __restrict__ bias,
    const float* __restrict__ add1, const float* __restrict__ add2,
    float* __restrict__ C, int ldc, int N, int mt, int ntp, int kbc, int gelu) {
  int lane = threadIdx.x & 31;
  int wave = threadIdx.x >> 5;
  int tn0 = (blockIdx.x * 2 + wave) * 2;
  int tm0 = blockIdx.y * 2;
  if (tn0 >= ntp) return;
  const _Float16* ap = Ap + ((size_t)tm0 * 32 + lane) * 16;
  const _Float16* bp = Bp + ((size_t)tn0 * 32 + lane) * 16;
  size_t aStep = (size_t)mt * 512;
  size_t bStep = (size_t)ntp * 512;
  v8f acc00 = {}, acc01 = {}, acc10 = {}, acc11 = {};
  v16h av0 = *(const v16h*)ap;
  v16h av1 = *(const v16h*)(ap + 512);
  v16h bv0 = *(const v16h*)bp;
  v16h bv1 = *(const v16h*)(bp + 512);
  for (int kb = 0; kb < kbc - 1; ++kb) {
    ap += aStep; bp += bStep;
    v16h nav0 = *(const v16h*)ap;
    v16h nav1 = *(const v16h*)(ap + 512);
    v16h nbv0 = *(const v16h*)bp;
    v16h nbv1 = *(const v16h*)(bp + 512);
    acc00 = wmma_f16(av0, bv0, acc00);
    acc01 = wmma_f16(av0, bv1, acc01);
    acc10 = wmma_f16(av1, bv0, acc10);
    acc11 = wmma_f16(av1, bv1, acc11);
    av0 = nav0; av1 = nav1; bv0 = nbv0; bv1 = nbv1;
  }
  acc00 = wmma_f16(av0, bv0, acc00);
  acc01 = wmma_f16(av0, bv1, acc01);
  acc10 = wmma_f16(av1, bv0, acc10);
  acc11 = wmma_f16(av1, bv1, acc11);

  int l15 = lane & 15, lhalf = lane >> 4;
#pragma unroll
  for (int ti = 0; ti < 2; ++ti) {
#pragma unroll
    for (int tj = 0; tj < 2; ++tj) {
      int bn = (tn0 + tj) * 16 + l15;
      if (bn >= N) continue;
      v8f a = ti ? (tj ? acc11 : acc10) : (tj ? acc01 : acc00);
      int m0 = (tm0 + ti) * 16;
#pragma unroll
      for (int r = 0; r < 8; ++r) {
        int cm = m0 + r + (lhalf << 3);
        float v = a[r];
        if (bias) v += bias[bn];
        if (add1) v += add1[(size_t)cm * ldc + bn];
        if (add2) v += add2[(size_t)cm * ldc + bn];
        if (gelu) v = gelu_exact(v);
        C[(size_t)cm * ldc + bn] = v;
      }
    }
  }
}

// ---------------------------------------------------------------- l2norm rows
__global__ void k_l2norm(float* __restrict__ q, int n, int d) {
  int t = blockIdx.x * blockDim.x + threadIdx.x;
  if (t >= n) return;
  float* row = q + (size_t)t * d;
  float s = 0.f;
  for (int i = 0; i < d; ++i) s += row[i] * row[i];
  float r = 1.f / fmaxf(sqrtf(s), 1e-12f);
  for (int i = 0; i < d; ++i) row[i] *= r;
}

// ---------------------------------------------------------------- dictionary k,v
__global__ void k_kk(const float* __restrict__ td, const float* __restrict__ wk,
                     const float* __restrict__ wkb, float* __restrict__ kkn) {
  int m = threadIdx.x;
  if (m >= M_TOK) return;
  float v[RC];
  for (int j = 0; j < RC; ++j) {
    float a = wkb[j];
    for (int k = 0; k < C_DIM; ++k) a += td[m * C_DIM + k] * wk[k * RC + j];
    v[j] = a;
  }
  float s = 0.f;
  for (int j = 0; j < RC; ++j) s += v[j] * v[j];
  float r = 1.f / fmaxf(sqrtf(s), 1e-12f);
  for (int j = 0; j < RC; ++j) kkn[m * RC + j] = v[j] * r;
}

__global__ void k_vtd(const float* __restrict__ td, const float* __restrict__ wv,
                      const float* __restrict__ wvb, float* __restrict__ vtd) {
  int idx = blockIdx.x * blockDim.x + threadIdx.x;
  if (idx >= M_TOK * C_DIM) return;
  int m = idx / C_DIM, ch = idx - m * C_DIM;
  float a = wvb[ch];
  for (int k = 0; k < C_DIM; ++k) a += td[m * C_DIM + k] * wv[k * C_DIM + ch];
  vtd[idx] = a;
}

// ---------------------------------------------------------------- sim softmax (per token)
__global__ __launch_bounds__(128) void k_sim(const float* __restrict__ qn,
    const float* __restrict__ kkn, const float* __restrict__ atd_scale,
    float* __restrict__ sim) {
  __shared__ float red[M_TOK];
  int t = blockIdx.x;
  int m = threadIdx.x;
  const float* q = qn + (size_t)t * RC;
  const float* kr = kkn + m * RC;
  float a = 0.f;
  for (int j = 0; j < RC; ++j) a += q[j] * kr[j];
  float sc = fminf(fmaxf(atd_scale[m], 0.f), 1.f);
  a *= (1.f + sc * LOG_M);
  red[m] = a; __syncthreads();
  for (int s = 64; s > 0; s >>= 1) { if (m < s) red[m] = fmaxf(red[m], red[m + s]); __syncthreads(); }
  float mx = red[0]; __syncthreads();
  float e = expf(a - mx);
  red[m] = e; __syncthreads();
  for (int s = 64; s > 0; s >>= 1) { if (m < s) red[m] += red[m + s]; __syncthreads(); }
  sim[(size_t)t * M_TOK + m] = e / red[0];
}

// ---------------------------------------------------------------- window attention
__device__ __forceinline__ int win_token_row(int win, int t) {
  int wy = win >> 4, wx = win & 15;
  int iy = t >> 4, ix = t & 15;
  int gy = (wy * WSZ + iy + SHIFT_) & (IMG - 1);
  int gx = (wx * WSZ + ix + SHIFT_) & (IMG - 1);
  return gy * IMG + gx;
}

// 4 waves/block; K,V staged in LDS f16 (shared), f16 score slab per wave. 64KB LDS.
__global__ __launch_bounds__(128) void k_attn(const float* __restrict__ qkv,
    const float* __restrict__ rpbt, const int* __restrict__ rpi,
    const float* __restrict__ mask, float* __restrict__ obuf) {
  __shared__ _Float16 ksh[NN_][HD];       // 16KB
  __shared__ _Float16 vsh[NN_][HD];       // 16KB
  __shared__ _Float16 sch[4][16][NN_];    // 32KB
  int rtg  = blockIdx.x;
  int win  = blockIdx.y;
  int head = blockIdx.z;
  int tid = threadIdx.x;
  int wave = tid >> 5, lane = tid & 31;
  int l15 = lane & 15, lhalf = lane >> 4;
  int rt = rtg * 4 + wave;

  for (int t = tid; t < NN_; t += 128) {
    const float* kr = qkv + (size_t)win_token_row(win, t) * C3 + C_DIM + head * HD;
#pragma unroll
    for (int d = 0; d < HD; d += 4) {
      float4 k4 = *(const float4*)(kr + d);
      ksh[t][d + 0] = (_Float16)k4.x; ksh[t][d + 1] = (_Float16)k4.y;
      ksh[t][d + 2] = (_Float16)k4.z; ksh[t][d + 3] = (_Float16)k4.w;
      float4 v4 = *(const float4*)(kr + C_DIM + d);
      vsh[t][d + 0] = (_Float16)v4.x; vsh[t][d + 1] = (_Float16)v4.y;
      vsh[t][d + 2] = (_Float16)v4.z; vsh[t][d + 3] = (_Float16)v4.w;
    }
  }
  __syncthreads();

  v16h av;
  {
    int qt = rt * 16 + l15;
    const float* qrow = qkv + (size_t)win_token_row(win, qt) * C3 + head * HD;
    int base = lhalf ? 16 : 0;
    float4 q0 = *(const float4*)(qrow + base);
    float4 q1 = *(const float4*)(qrow + base + 4);
    float4 q2 = *(const float4*)(qrow + base + 8);
    float4 q3 = *(const float4*)(qrow + base + 12);
    av[0] = (_Float16)(q0.x * QSCALE); av[1] = (_Float16)(q0.y * QSCALE);
    av[2] = (_Float16)(q0.z * QSCALE); av[3] = (_Float16)(q0.w * QSCALE);
    av[4] = (_Float16)(q1.x * QSCALE); av[5] = (_Float16)(q1.y * QSCALE);
    av[6] = (_Float16)(q1.z * QSCALE); av[7] = (_Float16)(q1.w * QSCALE);
    av[8] = (_Float16)(q2.x * QSCALE); av[9] = (_Float16)(q2.y * QSCALE);
    av[10] = (_Float16)(q2.z * QSCALE); av[11] = (_Float16)(q2.w * QSCALE);
    av[12] = (_Float16)(q3.x * QSCALE); av[13] = (_Float16)(q3.y * QSCALE);
    av[14] = (_Float16)(q3.z * QSCALE); av[15] = (_Float16)(q3.w * QSCALE);
  }
  for (int ct = 0; ct < 16; ++ct) {
    v16h bv = *(const v16h*)&ksh[ct * 16 + l15][lhalf ? 16 : 0];
    v8f a = {};
    a = wmma_f16(av, bv, a);
#pragma unroll
    for (int r = 0; r < 8; ++r)
      sch[wave][r + (lhalf << 3)][ct * 16 + l15] = (_Float16)a[r];
  }
  __syncthreads();
  {
    int row = lane >> 1;
    int c0 = (lane & 1) * 128;
    int gi = rt * 16 + row;
    const int* rpir = rpi + gi * NN_;
    const float* mrow = mask + ((size_t)win * NN_ + gi) * NN_;
    float mxl = -3.4e38f;
    for (int j = 0; j < 128; ++j) {
      int cc = c0 + j;
      float v = (float)sch[wave][row][cc] + rpbt[rpir[cc] * HEADS_ + head] + mrow[cc];
      sch[wave][row][cc] = (_Float16)v;
      mxl = fmaxf(mxl, v);
    }
    float mxr = fmaxf(mxl, __shfl_xor(mxl, 1, 32));
    float sml = 0.f;
    for (int j = 0; j < 128; ++j) {
      int cc = c0 + j;
      float p = expf((float)sch[wave][row][cc] - mxr);
      sch[wave][row][cc] = (_Float16)p;
      sml += p;
    }
    float z = sml + __shfl_xor(sml, 1, 32);
    float rs = 1.f / z;
    for (int j = 0; j < 128; ++j) {
      int cc = c0 + j;
      sch[wave][row][cc] = (_Float16)((float)sch[wave][row][cc] * rs);
    }
  }
  __syncthreads();
#pragma unroll
  for (int nt = 0; nt < 2; ++nt) {
    v8f a = {};
    for (int kc = 0; kc < 8; ++kc) {
      int base = (lhalf ? 8 : 0) + kc * 32;
      v8h lo = *(const v8h*)&sch[wave][l15][base];
      v8h hi = *(const v8h*)&sch[wave][l15][base + 16];
      v16h a2;
#pragma unroll
      for (int j = 0; j < 8; ++j) { a2[j] = lo[j]; a2[j + 8] = hi[j]; }
      v16h b2;
#pragma unroll
      for (int j = 0; j < 16; ++j) {
        int key = kc * 32 + (lhalf ? 8 : 0) + ((j < 8) ? j : (j + 8));
        b2[j] = vsh[key][nt * 16 + l15];
      }
      a = wmma_f16(a2, b2, a);
    }
#pragma unroll
    for (int r = 0; r < 8; ++r) {
      int qt = rt * 16 + r + (lhalf << 3);
      obuf[(size_t)win_token_row(win, qt) * C_DIM + head * HD + nt * 16 + l15] = a[r];
    }
  }
}

// ---------------------------------------------------------------- depthwise 5x5 + GELU + residual
__global__ void k_dwconv(const float* __restrict__ y, const float* __restrict__ w,
                         const float* __restrict__ bias, float* __restrict__ y2) {
  int idx = blockIdx.x * blockDim.x + threadIdx.x;
  if (idx >= N_TOK * HID) return;
  int t = idx / HID, ch = idx - t * HID;
  int ty = t >> 8, tx = t & 255;
  const float* wk = w + ch * 25;
  float a = bias[ch];
  for (int dy = 0; dy < 5; ++dy) {
    int yy = ty + dy - 2;
    if (yy < 0 || yy >= IMG) continue;
    for (int dx = 0; dx < 5; ++dx) {
      int xx = tx + dx - 2;
      if (xx < 0 || xx >= IMG) continue;
      a += y[((size_t)(yy * IMG + xx)) * HID + ch] * wk[dy * 5 + dx];
    }
  }
  y2[idx] = y[idx] + gelu_exact(a);
}

// ---------------------------------------------------------------- td-update stats
// st: [0..127]=inv, [128..255]=sh, [256..383]=max, [384..511]=1/Z
__global__ __launch_bounds__(256) void k_stats1(const float* __restrict__ sim,
    const float* __restrict__ g, const float* __restrict__ b, float* __restrict__ st) {
  __shared__ float s1[256], s2[256];
  int m = blockIdx.x, tid = threadIdx.x;
  float a = 0.f, aa = 0.f;
  for (int t = tid; t < N_TOK; t += 256) {
    float v = sim[(size_t)t * M_TOK + m];
    a += v; aa += v * v;
  }
  s1[tid] = a; s2[tid] = aa; __syncthreads();
  for (int s = 128; s > 0; s >>= 1) {
    if (tid < s) { s1[tid] += s1[tid + s]; s2[tid] += s2[tid + s]; }
    __syncthreads();
  }
  if (tid == 0) {
    float mu = s1[0] / (float)N_TOK;
    float var = s2[0] / (float)N_TOK - mu * mu;
    float inv = g[m] * rsqrtf(var + 1e-5f);
    st[m] = inv;
    st[M_TOK + m] = b[m] - mu * inv;
  }
}
__global__ __launch_bounds__(256) void k_stats2(const float* __restrict__ sim,
                                                float* __restrict__ st) {
  __shared__ float s1[256];
  int m = blockIdx.x, tid = threadIdx.x;
  float inv = st[m], sh = st[M_TOK + m];
  float mx = -3.4e38f;
  for (int t = tid; t < N_TOK; t += 256)
    mx = fmaxf(mx, sim[(size_t)t * M_TOK + m] * inv + sh);
  s1[tid] = mx; __syncthreads();
  for (int s = 128; s > 0; s >>= 1) {
    if (tid < s) s1[tid] = fmaxf(s1[tid], s1[tid + s]);
    __syncthreads();
  }
  if (tid == 0) st[2 * M_TOK + m] = s1[0];
}
__global__ __launch_bounds__(256) void k_stats3(const float* __restrict__ sim,
                                                float* __restrict__ st) {
  __shared__ float s1[256];
  int m = blockIdx.x, tid = threadIdx.x;
  float inv = st[m], sh = st[M_TOK + m], mx = st[2 * M_TOK + m];
  float a = 0.f;
  for (int t = tid; t < N_TOK; t += 256)
    a += expf(sim[(size_t)t * M_TOK + m] * inv + sh - mx);
  s1[tid] = a; __syncthreads();
  for (int s = 128; s > 0; s >>= 1) {
    if (tid < s) s1[tid] += s1[tid + s];
    __syncthreads();
  }
  if (tid == 0) st[3 * M_TOK + m] = 1.f / s1[0];
}

__global__ void k_zero(float* __restrict__ p, int count) {
  int i = blockIdx.x * blockDim.x + threadIdx.x;
  if (i < count) p[i] = 0.f;
}

// ---------------------------------------------------------------- split-K packed WMMA: acc += P @ x
// Wave: 2 M-tiles x 2 N-tiles, pipelined. grid (ntp/2, mt/2, 64 splits of 32 kb).
__global__ __launch_bounds__(32) void k_tdacc(const _Float16* __restrict__ Pp,
    const _Float16* __restrict__ xP, float* __restrict__ acc) {
  int lane = threadIdx.x;
  int tn0 = blockIdx.x * 2;
  int tm0 = blockIdx.y * 2;
  int ks = blockIdx.z;
  const int mt = M_TOK / 16, ntp = C_DIM / 16;
  const _Float16* ap = Pp + (((size_t)(ks * 32) * mt + tm0) * 32 + lane) * 16;
  const _Float16* bp = xP + (((size_t)(ks * 32) * ntp + tn0) * 32 + lane) * 16;
  size_t aStep = (size_t)mt * 512;
  size_t bStep = (size_t)ntp * 512;
  v8f acc00 = {}, acc01 = {}, acc10 = {}, acc11 = {};
  v16h av0 = *(const v16h*)ap;
  v16h av1 = *(const v16h*)(ap + 512);
  v16h bv0 = *(const v16h*)bp;
  v16h bv1 = *(const v16h*)(bp + 512);
  for (int kb = 0; kb < 31; ++kb) {
    ap += aStep; bp += bStep;
    v16h nav0 = *(const v16h*)ap;
    v16h nav1 = *(const v16h*)(ap + 512);
    v16h nbv0 = *(const v16h*)bp;
    v16h nbv1 = *(const v16h*)(bp + 512);
    acc00 = wmma_f16(av0, bv0, acc00);
    acc01 = wmma_f16(av0, bv1, acc01);
    acc10 = wmma_f16(av1, bv0, acc10);
    acc11 = wmma_f16(av1, bv1, acc11);
    av0 = nav0; av1 = nav1; bv0 = nbv0; bv1 = nbv1;
  }
  acc00 = wmma_f16(av0, bv0, acc00);
  acc01 = wmma_f16(av0, bv1, acc01);
  acc10 = wmma_f16(av1, bv0, acc10);
  acc11 = wmma_f16(av1, bv1, acc11);
  int l15 = lane & 15, lhalf = lane >> 4;
#pragma unroll
  for (int r = 0; r < 8; ++r) {
    int cm0 = tm0 * 16 + r + (lhalf << 3);
    int cm1 = cm0 + 16;
    atomicAdd(&acc[cm0 * C_DIM + tn0 * 16 + l15], acc00[r]);
    atomicAdd(&acc[cm0 * C_DIM + (tn0 + 1) * 16 + l15], acc01[r]);
    atomicAdd(&acc[cm1 * C_DIM + tn0 * 16 + l15], acc10[r]);
    atomicAdd(&acc[cm1 * C_DIM + (tn0 + 1) * 16 + l15], acc11[r]);
  }
}

// ---------------------------------------------------------------- final blend
__global__ void k_blend(const float* __restrict__ td, const float* __restrict__ sigma,
                        const float* __restrict__ st, const float* __restrict__ acc,
                        float* __restrict__ out) {
  int idx = blockIdx.x * blockDim.x + threadIdx.x;
  if (idx >= M_TOK * C_DIM) return;
  int m = idx / C_DIM;
  float s = 1.f / (1.f + expf(-sigma[m]));
  out[idx] = s * td[idx] + (1.f - s) * acc[idx] * st[3 * M_TOK + m];
}

// ---------------------------------------------------------------- host helpers
static inline void packA(const float* A, int M, int K, int lda, _Float16* out, hipStream_t s) {
  int mt = M / 16, kbc = K / 32;
  size_t total = (size_t)kbc * mt * 512;
  k_packA<<<(unsigned)((total + 255) / 256), 256, 0, s>>>(A, mt, kbc, lda, out);
}
static inline void packB(const float* B, int K, int N, int ntp, _Float16* out, hipStream_t s) {
  int kbc = K / 32;
  size_t total = (size_t)kbc * ntp * 512;
  k_packB<<<(unsigned)((total + 255) / 256), 256, 0, s>>>(B, kbc, N, ntp, out);
}
static inline void gemm(const _Float16* Ap, const _Float16* Bp, const float* bias,
    const float* add1, const float* add2, float* C, int ldc,
    int M, int N, int K, int gelu, hipStream_t s) {
  int mt = M / 16, ntp = (N + 15) / 16, kbc = K / 32;
  if (ntp & 1) ntp += 1;  // keep even (packB call sites already pad)
  dim3 g((ntp + 3) / 4, mt / 2, 1);
  k_gemm<<<g, 64, 0, s>>>(Ap, Bp, bias, add1, add2, C, ldc, N, mt, ntp, kbc, gelu);
}

extern "C" void kernel_launch(void* const* d_in, const int* in_sizes, int n_in,
                              void* d_out, int out_size, void* d_ws, size_t ws_size,
                              hipStream_t stream) {
  (void)in_sizes; (void)n_in; (void)out_size; (void)ws_size;
  const float* x      = (const float*)d_in[0];
  const float* td     = (const float*)d_in[1];
  const float* n1g    = (const float*)d_in[2];
  const float* n1b    = (const float*)d_in[3];
  const float* n2g    = (const float*)d_in[4];
  const float* n2b    = (const float*)d_in[5];
  const float* n3g    = (const float*)d_in[6];
  const float* n3b    = (const float*)d_in[7];
  const float* wqkv_w = (const float*)d_in[8];
  const float* wqkv_b = (const float*)d_in[9];
  const float* rpbt   = (const float*)d_in[10];
  const float* proj_w = (const float*)d_in[11];
  const float* proj_b = (const float*)d_in[12];
  const float* wq_w   = (const float*)d_in[13];
  const float* wq_b   = (const float*)d_in[14];
  const float* wk_w   = (const float*)d_in[15];
  const float* wk_b   = (const float*)d_in[16];
  const float* wv_w   = (const float*)d_in[17];
  const float* wv_b   = (const float*)d_in[18];
  const float* atd_s  = (const float*)d_in[19];
  const float* sigma  = (const float*)d_in[20];
  const float* fc1_w  = (const float*)d_in[21];
  const float* fc1_b  = (const float*)d_in[22];
  const float* dw_w   = (const float*)d_in[23];
  const float* dw_b   = (const float*)d_in[24];
  const float* fc2_w  = (const float*)d_in[25];
  const float* fc2_b  = (const float*)d_in[26];
  const float* amask  = (const float*)d_in[27];
  const int*   rpi    = (const int*)d_in[28];

  char* ws = (char*)d_ws;
  size_t off = 0;
  float* qkv  = (float*)(ws + off); off += (size_t)N_TOK * C3 * 4;     // later: y
  float* sim  = (float*)(ws + off); off += (size_t)N_TOK * M_TOK * 4;
  float* xn   = (float*)(ws + off); off += (size_t)N_TOK * C_DIM * 4;  // later: x1
  float* atd  = (float*)(ws + off); off += (size_t)N_TOK * C_DIM * 4;
  float* obuf = (float*)(ws + off); off += (size_t)N_TOK * C_DIM * 4;  // later: xn2
  float* y2   = (float*)(ws + off); off += (size_t)N_TOK * HID * 4;
  float* qn   = (float*)(ws + off); off += (size_t)N_TOK * RC * 4;
  float* kkn  = (float*)(ws + off); off += (size_t)M_TOK * RC * 4;
  float* vtd  = (float*)(ws + off); off += (size_t)M_TOK * C_DIM * 4;
  float* st   = (float*)(ws + off); off += (size_t)4 * M_TOK * 4;
  float* acc  = (float*)(ws + off); off += (size_t)M_TOK * C_DIM * 4;
  _Float16* arena = (_Float16*)(ws + off); off += (size_t)N_TOK * HID * 2;
  _Float16* xP    = (_Float16*)(ws + off); off += (size_t)N_TOK * C_DIM * 2;
  _Float16* wqkvP = (_Float16*)(ws + off); off += (size_t)6 * 36 * 512 * 2;
  _Float16* wqP   = (_Float16*)(ws + off); off += (size_t)6 * 2 * 512 * 2;
  _Float16* projP = (_Float16*)(ws + off); off += (size_t)6 * 12 * 512 * 2;
  _Float16* fc1P  = (_Float16*)(ws + off); off += (size_t)6 * 24 * 512 * 2;
  _Float16* fc2P  = (_Float16*)(ws + off); off += (size_t)12 * 12 * 512 * 2;
  _Float16* vtdP  = (_Float16*)(ws + off); off += (size_t)4 * 12 * 512 * 2;
  float* y   = qkv;
  float* x1  = xn;
  float* xn2 = obuf;

  float* xout  = (float*)d_out;
  float* tdout = xout + (size_t)N_TOK * C_DIM;

  // pack static weights
  packB(wqkv_w, C_DIM, C3, 36, wqkvP, stream);
  packB(wq_w, C_DIM, RC, 2, wqP, stream);
  packB(proj_w, C_DIM, C_DIM, 12, projP, stream);
  packB(fc1_w, C_DIM, HID, 24, fc1P, stream);
  packB(fc2_w, HID, C_DIM, 12, fc2P, stream);

  // 1. LN1, pack
  k_layernorm<<<N_TOK / 8, 256, 0, stream>>>(x, n1g, n1b, xn, N_TOK, C_DIM);
  packA(xn, N_TOK, C_DIM, C_DIM, arena, stream);
  // 2. qkv
  gemm(arena, wqkvP, wqkv_b, nullptr, nullptr, qkv, C3, N_TOK, C3, C_DIM, 0, stream);
  // 3. q + l2norm
  gemm(arena, wqP, wq_b, nullptr, nullptr, qn, RC, N_TOK, RC, C_DIM, 0, stream);
  k_l2norm<<<N_TOK / 256, 256, 0, stream>>>(qn, N_TOK, RC);
  // 4. dictionary k/v
  k_kk<<<1, 128, 0, stream>>>(td, wk_w, wk_b, kkn);
  k_vtd<<<(M_TOK * C_DIM + 255) / 256, 256, 0, stream>>>(td, wv_w, wv_b, vtd);
  packB(vtd, M_TOK, C_DIM, 12, vtdP, stream);
  // 5. sim
  k_sim<<<N_TOK, 128, 0, stream>>>(qn, kkn, atd_s, sim);
  // 6. x_atd = sim @ v_td
  packA(sim, N_TOK, M_TOK, M_TOK, arena, stream);
  gemm(arena, vtdP, nullptr, nullptr, nullptr, atd, C_DIM, N_TOK, C_DIM, M_TOK, 0, stream);
  // 7. window attention
  {
    dim3 g(4, 256, HEADS_);
    k_attn<<<g, 128, 0, stream>>>(qkv, rpbt, rpi, amask, obuf);
  }
  // 8. x1 = x + atd + obuf @ proj + b
  packA(obuf, N_TOK, C_DIM, C_DIM, arena, stream);
  gemm(arena, projP, proj_b, x, atd, x1, C_DIM, N_TOK, C_DIM, C_DIM, 0, stream);
  // 9. LN2
  k_layernorm<<<N_TOK / 8, 256, 0, stream>>>(x1, n2g, n2b, xn2, N_TOK, C_DIM);
  // 10. y = gelu(xn2 @ fc1 + b)
  packA(xn2, N_TOK, C_DIM, C_DIM, arena, stream);
  gemm(arena, fc1P, fc1_b, nullptr, nullptr, y, HID, N_TOK, HID, C_DIM, 1, stream);
  // 11. y2 = y + gelu(dwconv(y)+b)
  k_dwconv<<<(N_TOK * HID + 255) / 256, 256, 0, stream>>>(y, dw_w, dw_b, y2);
  // 12. xout = x1 + y2 @ fc2 + b
  packA(y2, N_TOK, HID, HID, arena, stream);
  gemm(arena, fc2P, fc2_b, x1, nullptr, xout, C_DIM, N_TOK, C_DIM, HID, 0, stream);
  // 13. stats
  k_stats1<<<M_TOK, 256, 0, stream>>>(sim, n3g, n3b, st);
  k_stats2<<<M_TOK, 256, 0, stream>>>(sim, st);
  k_stats3<<<M_TOK, 256, 0, stream>>>(sim, st);
  // 14. packed P and x, split-K accumulation
  k_packP<<<(unsigned)(((size_t)2048 * 8 * 512 + 255) / 256), 256, 0, stream>>>(sim, st, arena);
  packB(xout, N_TOK, C_DIM, 12, xP, stream);
  k_zero<<<(M_TOK * C_DIM + 255) / 256, 256, 0, stream>>>(acc, M_TOK * C_DIM);
  {
    dim3 g(C_DIM / 32, M_TOK / 32, 64);   // (6, 4, 64)
    k_tdacc<<<g, 32, 0, stream>>>(arena, xP, acc);
  }
  // 15. blend
  k_blend<<<(M_TOK * C_DIM + 255) / 256, 256, 0, stream>>>(td, sigma, st, acc, tdout);
}
